// TaskAwareRelation_506806141136
// MI455X (gfx1250) — compile-verified, matrix operand order checked by reading, not run on hardware
//
#include <hip/hip_runtime.h>

typedef _Float16 half_t;
typedef __attribute__((ext_vector_type(16))) _Float16 v16h;
typedef __attribute__((ext_vector_type(8)))  _Float16 v8h;
typedef __attribute__((ext_vector_type(4)))  _Float16 v4h;
typedef __attribute__((ext_vector_type(8)))  float    v8f;
typedef __attribute__((ext_vector_type(4)))  float    v4f;

__device__ __forceinline__ int lane_id() { return threadIdx.x & 31; }

// leaky_relu(x, 0.01) == max(x, 0.01*x) for positive slope: mul+max, no VCC chain.
__device__ __forceinline__ float leaky(float x) { return fmaxf(x, 0.01f * x); }

// Load a 16x32 (f16) WMMA A/B fragment from a row-major [16+][ld] matrix.
// Row = lane%16 (relative to base), K window = kt*32.
// wave32 layout: lanes 0-15 hold K {0..7,16..23}; lanes 16-31 hold K {8..15,24..31}.
__device__ __forceinline__ v16h load_frag(const half_t* __restrict__ base, int ld, int kt) {
  const int l = lane_id();
  const half_t* p = base + (long)(l & 15) * ld + kt * 32 + ((l >> 4) & 1) * 8;
  v8h lo = *(const v8h*)(p);
  v8h hi = *(const v8h*)(p + 16);
  v16h r;
#pragma unroll
  for (int t = 0; t < 8; ++t) { r[t] = lo[t]; r[t + 8] = hi[t]; }
  return r;
}

__device__ __forceinline__ v8f wmma_f16(v16h a, v16h b, v8f c) {
  return __builtin_amdgcn_wmma_f32_16x16x32_f16(false, a, false, b, (short)0, c, false, false);
}

// Store a 16x16 f32 D fragment to row-major half [16][LD] (LDS) with leaky-relu.
// C/D layout: lanes 0-15 -> N=lane, M=r; lanes 16-31 -> N=lane-16, M=r+8.
template <int LD>
__device__ __forceinline__ void store_frag_h(half_t* dst, int nt, v8f c) {
  const int l = lane_id();
  const int n = nt * 16 + (l & 15);
  const int mb = (l >> 4) * 8;
#pragma unroll
  for (int r = 0; r < 8; ++r)
    dst[(mb + r) * LD + n] = (half_t)leaky(c[r]);
}

// One MLP layer on two M-tiles: dst[32][NOUT] = leaky(src[32][KD] @ W[NOUT][KD]^T).
// B fragments double-buffered; each B frag feeds both M-tiles.
template <int KD, int NOUT>
__device__ __forceinline__ void mlp_layer(const half_t* src, half_t* dst,
                                          const half_t* __restrict__ W) {
  constexpr int KT = KD / 32;
#pragma unroll
  for (int nt = 0; nt < NOUT / 16; ++nt) {
    const half_t* wbase = W + nt * 16 * KD;
    v8f acc0 = {}, acc1 = {};
    v16h cur = load_frag(wbase, KD, 0);
#pragma unroll
    for (int kt = 0; kt < KT; ++kt) {
      v16h nxt = cur;
      if (kt + 1 < KT) nxt = load_frag(wbase, KD, kt + 1);
      acc0 = wmma_f16(load_frag(src, KD, kt), cur, acc0);
      acc1 = wmma_f16(load_frag(src + 16 * KD, KD, kt), cur, acc1);
      cur = nxt;
    }
    store_frag_h<NOUT>(dst, nt, acc0);
    store_frag_h<NOUT>(dst + 16 * NOUT, nt, acc1);
  }
}

// ---------------------------------------------------------------------------
// Fused edge MLP: s[b,i,j] = MLP(exp(-|node[b,i,:]-node[b,j,:]|)).
// Block = 64 threads = 2 waves; each wave owns 2 M-tiles (32 pairs, rows 0-31
// of its private LDS ping-pong buffers). Block covers 64 j's of one (b,i).
// ---------------------------------------------------------------------------
template <int IN>
__global__ __launch_bounds__(64) void edge_mlp_kernel(
    const float* __restrict__ node, int node_ld,
    const half_t* __restrict__ w0,   // [256][IN]
    const half_t* __restrict__ w1,   // [128][256]
    const half_t* __restrict__ w2,   // [128][128]
    const half_t* __restrict__ wo,   // [128]
    const float* __restrict__ bo_ptr,
    float* __restrict__ s_out)       // [16][128][128]
{
  constexpr int H = 128, H2 = 256;
  __shared__ half_t ldsA[2][32 * IN];   // x, then layer-1 out (32x128 fits: IN>=128)
  __shared__ half_t ldsB[2][32 * H2];   // layer-0 out, then layer-2 out
  const int wave = threadIdx.x >> 5;
  const int l = threadIdx.x & 31;
  half_t* bufA = &ldsA[wave][0];
  half_t* bufB = &ldsB[wave][0];

  const int pair0 = blockIdx.x * 64;
  const int b  = pair0 >> 14;          // / (128*128)
  const int i  = (pair0 >> 7) & 127;
  const int j0 = (pair0 & 127) + wave * 32;

  // Build x = exp(-|n_i - n_j|) into bufA as row-major [32][IN] f16; lane = row.
  {
    const int j = j0 + l;
    const float* pi = node + (long)(b * 128 + i) * node_ld;
    const float* pj = node + (long)(b * 128 + j) * node_ld;
#pragma unroll
    for (int c = 0; c < IN; c += 4) {
      v4f va = *(const v4f*)(pi + c);
      v4f vb = *(const v4f*)(pj + c);
      v4h o;
#pragma unroll
      for (int t = 0; t < 4; ++t) o[t] = (half_t)__expf(-fabsf(va[t] - vb[t]));
      *(v4h*)(bufA + l * IN + c) = o;
    }
  }

  mlp_layer<IN, H2>(bufA, bufB, w0);   // x -> h0 [32][256]
  mlp_layer<H2, H>(bufB, bufA, w1);    // h0 -> h1 [32][128]
  mlp_layer<H, H>(bufA, bufB, w2);     // h1 -> h2 [32][128]

  // Output: s = h2 @ wo + bo.  ld=0 trick: every lane reads wo's row so all D
  // columns equal the score; lanes 0/16 own column 0.
  {
    v16h bfr[H / 32];
#pragma unroll
    for (int kt = 0; kt < H / 32; ++kt) bfr[kt] = load_frag(wo, 0, kt);
    v8f acc0 = {}, acc1 = {};
#pragma unroll
    for (int kt = 0; kt < H / 32; ++kt) {
      acc0 = wmma_f16(load_frag(bufB, H, kt), bfr[kt], acc0);
      acc1 = wmma_f16(load_frag(bufB + 16 * H, H, kt), bfr[kt], acc1);
    }
    if ((l & 15) == 0) {
      const float bo = bo_ptr[0];
      const int mb = (l >> 4) * 8;
      float* sp = s_out + (long)(b * 128 + i) * 128 + j0;
#pragma unroll
      for (int r = 0; r < 8; ++r) {
        sp[mb + r] = acc0[r] + bo;
        sp[16 + mb + r] = acc1[r] + bo;
      }
    }
  }
}

// ---------------------------------------------------------------------------
// Softmax / dsim / L1-normalized e.  One block per (b,i) row, 128 threads.
// ---------------------------------------------------------------------------
__device__ __forceinline__ float block_max(float v, float* red) {
  red[threadIdx.x] = v; __syncthreads();
  for (int s = 64; s > 0; s >>= 1) {
    if ((int)threadIdx.x < s) red[threadIdx.x] = fmaxf(red[threadIdx.x], red[threadIdx.x + s]);
    __syncthreads();
  }
  float r = red[0]; __syncthreads(); return r;
}
__device__ __forceinline__ float block_sum(float v, float* red) {
  red[threadIdx.x] = v; __syncthreads();
  for (int s = 64; s > 0; s >>= 1) {
    if ((int)threadIdx.x < s) red[threadIdx.x] += red[threadIdx.x + s];
    __syncthreads();
  }
  float r = red[0]; __syncthreads(); return r;
}

__global__ __launch_bounds__(128) void softmax_kernel(
    const float* __restrict__ s,
    float* __restrict__ adj,   // [16][2][128][128] (section of d_out)
    half_t* __restrict__ e)    // [16][2][128][128] f16, L1-normalized rows
{
  __shared__ float red[128];
  const int b = blockIdx.x >> 7;
  const int i = blockIdx.x & 127;
  const int j = threadIdx.x;
  const float diag = (j == i) ? 1.0f : 0.0f;

  float x = s[(long)(b * 128 + i) * 128 + j] - diag * 1e8f;
  float m = block_max(x, red);
  float p = __expf(x - m);
  float sum = block_sum(p, red);
  float sim = p / sum;
  adj[((long)(b * 2 + 0) * 128 + i) * 128 + j] = sim;
  float e0 = (j == i) ? 0.0f : sim;
  float l1 = block_sum(fabsf(e0), red);
  e[((long)(b * 2 + 0) * 128 + i) * 128 + j] = (half_t)(e0 / fmaxf(l1, 1e-12f));

  float y = (1.0f - sim) - diag * 1e8f;
  float m2 = block_max(y, red);
  float p2 = __expf(y - m2);
  float sum2 = block_sum(p2, red);
  float dsim = p2 / sum2;
  adj[((long)(b * 2 + 1) * 128 + i) * 128 + j] = dsim;
  float e1 = (j == i) ? 0.0f : dsim;
  float l12 = block_sum(fabsf(e1), red);
  e[((long)(b * 2 + 1) * 128 + i) * 128 + j] = (half_t)(e1 / fmaxf(l12, 1e-12f));
}

// node f32 -> f16 copy into feat[:, 0:DIN] and transposed nodeT[b][d][j].
__global__ __launch_bounds__(128) void prep_kernel(
    const float* __restrict__ node, int node_ld, int DIN,
    half_t* __restrict__ feat, half_t* __restrict__ nodeT)
{
  const int b = blockIdx.x >> 7;
  const int i = blockIdx.x & 127;
  const int fld = 3 * DIN;
  for (int c = threadIdx.x; c < DIN; c += 128) {
    half_t h = (half_t)node[(long)(b * 128 + i) * node_ld + c];
    feat[(long)(b * 128 + i) * fld + c] = h;
    nodeT[(long)(b * DIN + c) * 128 + i] = h;
  }
}

// aggr[b,ch,i,:] = e[b,ch,i,:] @ node[b]  -> feat[:, (1+ch)*DIN : (2+ch)*DIN]
// D staged through LDS, then coalesced 16B stores.
template <int DIN>
__global__ __launch_bounds__(128) void aggr_kernel(
    const half_t* __restrict__ e, const half_t* __restrict__ nodeT,
    half_t* __restrict__ feat)
{
  __shared__ half_t stg[4][16 * 16];
  const int wave = threadIdx.x >> 5;
  const int l = threadIdx.x & 31;
  const int wu = blockIdx.x * 4 + wave;  // 0..255
  const int b  = wu >> 4;
  const int ch = (wu >> 3) & 1;
  const int it = wu & 7;
  const half_t* A  = e + ((long)(b * 2 + ch) * 128 + it * 16) * 128;
  const half_t* Bp = nodeT + (long)b * DIN * 128;
  constexpr int fld = 3 * DIN;
  half_t* st = stg[wave];
  for (int nt = 0; nt < DIN / 16; ++nt) {
    const half_t* wbase = Bp + nt * 16 * 128;
    v16h bfr[4];
#pragma unroll
    for (int kt = 0; kt < 4; ++kt) bfr[kt] = load_frag(wbase, 128, kt);
    v8f acc = {};
#pragma unroll
    for (int kt = 0; kt < 4; ++kt)
      acc = wmma_f16(load_frag(A, 128, kt), bfr[kt], acc);
    {
      const int n = l & 15, mb = (l >> 4) * 8;
#pragma unroll
      for (int r = 0; r < 8; ++r) st[(mb + r) * 16 + n] = (half_t)acc[r];
    }
    {
      const int row = l >> 1, cc = (l & 1) * 8;
      v8h v = *(const v8h*)(st + row * 16 + cc);
      *(v8h*)(feat + (long)(b * 128 + it * 16 + row) * fld + (1 + ch) * DIN + nt * 16 + cc) = v;
    }
  }
}

// out[:, col0:col0+128] = leaky(X[2048][K] @ W[128][K]^T + bias).
// D staged through LDS, then coalesced b128 stores.
template <int K>
__global__ __launch_bounds__(128) void gemm_h_kernel(
    const half_t* __restrict__ X,
    const half_t* __restrict__ W,   // [128][K]
    const float* __restrict__ bias,
    float* __restrict__ out, int out_ld, int out_col0)
{
  __shared__ float stg[4][16 * 16];
  const int wave = threadIdx.x >> 5;
  const int l = threadIdx.x & 31;
  const int wu = blockIdx.x * 4 + wave;  // 0..127 row-tiles
  const half_t* A = X + (long)wu * 16 * K;
  float* st = stg[wave];
  constexpr int KT = K / 32;
  for (int nt = 0; nt < 8; ++nt) {
    const half_t* wbase = W + nt * 16 * K;
    v8f acc = {};
    v16h cur = load_frag(wbase, K, 0);
#pragma unroll
    for (int kt = 0; kt < KT; ++kt) {
      v16h nxt = cur;
      if (kt + 1 < KT) nxt = load_frag(wbase, K, kt + 1);
      acc = wmma_f16(load_frag(A, K, kt), cur, acc);
      cur = nxt;
    }
    {
      const int n = l & 15, mb = (l >> 4) * 8;
      const float bb = bias ? bias[nt * 16 + n] : 0.0f;
#pragma unroll
      for (int r = 0; r < 8; ++r) st[(mb + r) * 16 + n] = leaky(acc[r] + bb);
    }
    {
      const int row = l >> 1, cc = (l & 1) * 8;
      float* op = out + (long)(wu * 16 + row) * out_ld + out_col0 + nt * 16 + cc;
      *(v4f*)(op)     = *(const v4f*)(st + row * 16 + cc);
      *(v4f*)(op + 4) = *(const v4f*)(st + row * 16 + cc + 4);
    }
  }
}

__global__ void convert_f32_to_f16(const float* __restrict__ src, half_t* __restrict__ dst, int n) {
  int i = blockIdx.x * 256 + threadIdx.x;
  if (i < n) dst[i] = (half_t)src[i];
}

__global__ void copy_emb_kernel(const float* __restrict__ src, float* __restrict__ node_cat) {
  int idx = blockIdx.x * 256 + threadIdx.x;
  if (idx < 16 * 128 * 128) node_cat[(long)(idx >> 7) * 384 + (idx & 127)] = src[idx];
}

// s_logits (127x2) + q_logits (16x2)
__global__ void head_kernel(const float* __restrict__ node3,
                            const float* __restrict__ fc2_w,
                            const float* __restrict__ fc2_b,
                            float* __restrict__ out)
{
  int t = blockIdx.x * blockDim.x + threadIdx.x;
  if (t < 127) {
    float a0 = 0.f, a1 = 0.f;
    for (int d = 0; d < 128; ++d) {
      float m = 0.f;
      for (int b = 0; b < 16; ++b) m += node3[(long)(b * 128 + t) * 128 + d];
      m *= (1.0f / 16.0f);
      a0 += m * fc2_w[d];
      a1 += m * fc2_w[128 + d];
    }
    out[t * 2 + 0] = a0 + fc2_b[0];
    out[t * 2 + 1] = a1 + fc2_b[1];
  } else if (t < 127 + 16) {
    int b = t - 127;
    const float* q = node3 + (long)(b * 128 + 127) * 128;
    float a0 = 0.f, a1 = 0.f;
    for (int d = 0; d < 128; ++d) { a0 += q[d] * fc2_w[d]; a1 += q[d] * fc2_w[128 + d]; }
    out[254 + b * 2 + 0] = a0 + fc2_b[0];
    out[254 + b * 2 + 1] = a1 + fc2_b[1];
  }
}

extern "C" void kernel_launch(void* const* d_in, const int* in_sizes, int n_in,
                              void* d_out, int out_size, void* d_ws, size_t ws_size,
                              hipStream_t stream) {
  (void)in_sizes; (void)n_in; (void)out_size; (void)ws_size;
  const float* all_emb = (const float*)d_in[0];
  float* out = (float*)d_out;

  char* ws = (char*)d_ws;
  size_t off = 0;
  auto alloc = [&](size_t bytes) -> char* {
    char* p = ws + off;
    off = (off + bytes + 255) & ~(size_t)255;
    return p;
  };

  // f16 weight copies
  half_t* e0w0h = (half_t*)alloc((size_t)256 * 128 * 2);
  half_t* e0w1h = (half_t*)alloc((size_t)128 * 256 * 2);
  half_t* e0w2h = (half_t*)alloc((size_t)128 * 128 * 2);
  half_t* e0woh = (half_t*)alloc((size_t)128 * 2);
  half_t* n0wh  = (half_t*)alloc((size_t)128 * 384 * 2);
  half_t* e1w0h = (half_t*)alloc((size_t)256 * 256 * 2);
  half_t* e1w1h = (half_t*)alloc((size_t)128 * 256 * 2);
  half_t* e1w2h = (half_t*)alloc((size_t)128 * 128 * 2);
  half_t* e1woh = (half_t*)alloc((size_t)128 * 2);
  half_t* n1wh  = (half_t*)alloc((size_t)128 * 768 * 2);
  half_t* fc1wh = (half_t*)alloc((size_t)128 * 384 * 2);
  // buffers
  float*  s_buf    = (float*)alloc((size_t)16 * 128 * 128 * 4);      // scores, reused as node3
  half_t* e_buf    = (half_t*)alloc((size_t)16 * 2 * 128 * 128 * 2); // normalized adj (f16)
  float*  node_cat = (float*)alloc((size_t)2048 * 384 * 4);          // growing node features
  half_t* nodeT    = (half_t*)alloc((size_t)16 * 256 * 128 * 2);     // node transposed (f16)
  half_t* feat     = (half_t*)alloc((size_t)2048 * 768 * 2);         // [node|aggr0|aggr1] (f16)
  float*  node3    = s_buf;   // reuse (scores dead after softmax1)
  half_t* node2h   = feat;    // reuse (feat dead after stage1 node-MLP)

  auto cvt = [&](const void* src, half_t* dst, int n) {
    convert_f32_to_f16<<<(n + 255) / 256, 256, 0, stream>>>((const float*)src, dst, n);
  };
  cvt(d_in[1], e0w0h, 256 * 128);
  cvt(d_in[2], e0w1h, 128 * 256);
  cvt(d_in[3], e0w2h, 128 * 128);
  cvt(d_in[4], e0woh, 128);
  cvt(d_in[6], n0wh, 128 * 384);
  cvt(d_in[7], e1w0h, 256 * 256);
  cvt(d_in[8], e1w1h, 128 * 256);
  cvt(d_in[9], e1w2h, 128 * 128);
  cvt(d_in[10], e1woh, 128);
  cvt(d_in[12], n1wh, 128 * 768);
  cvt(d_in[13], fc1wh, 128 * 384);

  copy_emb_kernel<<<(16 * 128 * 128 + 255) / 256, 256, 0, stream>>>(all_emb, node_cat);

  const long adj0_off = 254 + 32;
  const long adj1_off = adj0_off + (long)16 * 2 * 128 * 128;

  // ---- stage 0 (in = 128) ----
  edge_mlp_kernel<128><<<4096, 64, 0, stream>>>(node_cat, 384, e0w0h, e0w1h, e0w2h,
                                                e0woh, (const float*)d_in[5], s_buf);
  softmax_kernel<<<2048, 128, 0, stream>>>(s_buf, out + adj0_off, e_buf);
  prep_kernel<<<2048, 128, 0, stream>>>(node_cat, 384, 128, feat, nodeT);
  aggr_kernel<128><<<64, 128, 0, stream>>>(e_buf, nodeT, feat);
  gemm_h_kernel<384><<<32, 128, 0, stream>>>(feat, n0wh, nullptr, node_cat, 384, 128);

  // ---- stage 1 (in = 256) ----
  edge_mlp_kernel<256><<<4096, 64, 0, stream>>>(node_cat, 384, e1w0h, e1w1h, e1w2h,
                                                e1woh, (const float*)d_in[11], s_buf);
  softmax_kernel<<<2048, 128, 0, stream>>>(s_buf, out + adj1_off, e_buf);
  prep_kernel<<<2048, 128, 0, stream>>>(node_cat, 384, 256, feat, nodeT);
  aggr_kernel<256><<<64, 128, 0, stream>>>(e_buf, nodeT, feat);
  gemm_h_kernel<768><<<32, 128, 0, stream>>>(feat, n1wh, nullptr, node_cat, 384, 256);

  // ---- head: fc1 (384 -> 128, leaky) then fc2 logits ----
  convert_f32_to_f16<<<(2048 * 384 + 255) / 256, 256, 0, stream>>>(node_cat, node2h, 2048 * 384);
  gemm_h_kernel<384><<<32, 128, 0, stream>>>(node2h, fc1wh,
                                             (const float*)d_in[14], node3, 128, 0);
  head_kernel<<<1, 160, 0, stream>>>(node3, (const float*)d_in[15], (const float*)d_in[16], out);
}
